// BPTTNonLRTUs_8993661518096
// MI455X (gfx1250) — matrix operand order repaired and checked by reading, not compiled, not run on hardware
//
#include <hip/hip_runtime.h>

// ---------------------------------------------------------------------------
// Problem constants (reference: T=2048, B=64, D=256, H=256)
// ---------------------------------------------------------------------------
#define TT 2048
#define BB 64
#define DD 256
#define HH 256
#define NTOT (2 * HH)        // 512: concatenated [Wx1 | Wx2] output columns
#define MTOT (TT * BB)       // 131072 rows of xs

#define CHUNK_T 512          // timesteps per GEMM/scan chunk (ys chunk 64MB -> L2)
#define NCHUNK (TT / CHUNK_T)
#define MBLK 256             // GEMM block tile M
#define PF 16                // scan prefetch group depth (timesteps)

typedef _Float16 v16h __attribute__((ext_vector_type(16)));
typedef _Float16 v8h  __attribute__((ext_vector_type(8)));
typedef float    v8f  __attribute__((ext_vector_type(8)));
typedef float    v4f  __attribute__((ext_vector_type(4)));

// LDS row stride (halves) for the transposed weight tile: 256 + 8 pad
#define LDS_STRIDE 264

// ---------------------------------------------------------------------------
// Pass 1 (per T-chunk): C[m,0:256]=X@Wx1 ; C[m,256:512]=X@Wx2 into ys region.
// Block: 256 threads = 8 waves. Block tile: 256 (M) x 64 (N). K = 256.
// Wave w owns M-subtiles w and w+8 (rows w*16 and w*16+128): each B fragment
// feeds TWO wmma ops. 8 k-steps x 8 wmma = 64 static v_wmma per block.
// ---------------------------------------------------------------------------
__global__ __launch_bounds__(256)
void proj_gemm_kernel(const float* __restrict__ xs,
                      const float* __restrict__ Wx1,
                      const float* __restrict__ Wx2,
                      float* __restrict__ out, int mbase) {
  __shared__ _Float16 Wlds[64 * LDS_STRIDE];

  const int n0 = blockIdx.x * 64;                 // never straddles 256
  const int m0 = mbase + blockIdx.y * MBLK;
  const float* __restrict__ Wsrc = (n0 < HH) ? Wx1 : Wx2;
  const int colbase = n0 & (HH - 1);

  // Stage W[k][colbase+nl] -> Wlds[nl*LDS_STRIDE + k] as f16 (K-contiguous).
  for (int i = threadIdx.x; i < 64 * 256; i += 256) {
    const int k  = i >> 6;
    const int nl = i & 63;
    Wlds[nl * LDS_STRIDE + k] = (_Float16)Wsrc[k * HH + colbase + nl];
  }
  __syncthreads();

  const int lane  = threadIdx.x & 31;
  const int wv    = threadIdx.x >> 5;
  const int lhalf = lane >> 4;     // 0: lanes 0-15, 1: lanes 16-31
  const int lmod  = lane & 15;

  const float* __restrict__ xrow0 = xs + (size_t)(m0 + wv * 16 + lmod) * DD;
  const float* __restrict__ xrow1 = xrow0 + (size_t)128 * DD;   // subtile w+8

  v8f acc[2][4] = {{{}, {}, {}, {}}, {{}, {}, {}, {}}};

  #pragma unroll
  for (int k0 = 0; k0 < DD; k0 += 32) {
    // A fragments (16-bit A 16x32 layout): lanes 0-15 halves = K k0+0..7 /
    // k0+16..23 ; lanes 16-31 halves = K k0+8..15 / k0+24..31. NT: read-once.
    v16h a0, a1;
    {
      const float* p = xrow0 + k0 + (lhalf << 3);
      v4f q0 = __builtin_nontemporal_load((const v4f*)(p));
      v4f q1 = __builtin_nontemporal_load((const v4f*)(p + 4));
      v4f q2 = __builtin_nontemporal_load((const v4f*)(p + 16));
      v4f q3 = __builtin_nontemporal_load((const v4f*)(p + 20));
      #pragma unroll
      for (int j = 0; j < 4; ++j) {
        a0[j] = (_Float16)q0[j];  a0[4 + j]  = (_Float16)q1[j];
        a0[8 + j] = (_Float16)q2[j]; a0[12 + j] = (_Float16)q3[j];
      }
    }
    {
      const float* p = xrow1 + k0 + (lhalf << 3);
      v4f q0 = __builtin_nontemporal_load((const v4f*)(p));
      v4f q1 = __builtin_nontemporal_load((const v4f*)(p + 4));
      v4f q2 = __builtin_nontemporal_load((const v4f*)(p + 16));
      v4f q3 = __builtin_nontemporal_load((const v4f*)(p + 20));
      #pragma unroll
      for (int j = 0; j < 4; ++j) {
        a1[j] = (_Float16)q0[j];  a1[4 + j]  = (_Float16)q1[j];
        a1[8 + j] = (_Float16)q2[j]; a1[12 + j] = (_Float16)q3[j];
      }
    }

    // B fragments: lane n holds column n, halves = K kb..kb+7 / kb+16..kb+23.
    const int kb = k0 + (lhalf << 3);
    #pragma unroll
    for (int ns = 0; ns < 4; ++ns) {
      const _Float16* bp = &Wlds[(ns * 16 + lmod) * LDS_STRIDE + kb];
      v8h b0 = *(const v8h*)(bp);
      v8h b1 = *(const v8h*)(bp + 16);
      v16h b;
      #pragma unroll
      for (int j = 0; j < 8; ++j) { b[j] = b0[j]; b[8 + j] = b1[j]; }
      acc[0][ns] = __builtin_amdgcn_wmma_f32_16x16x32_f16(
          false, a0, false, b, (short)0, acc[0][ns], false, false);
      acc[1][ns] = __builtin_amdgcn_wmma_f32_16x16x32_f16(
          false, a1, false, b, (short)0, acc[1][ns], false, false);
    }
  }

  // Store (RT: keep resident in L2 for the scan chunk that follows).
  const int ncol = n0 + lmod;
  #pragma unroll
  for (int part = 0; part < 2; ++part) {
    const size_t rbase = (size_t)(m0 + wv * 16 + part * 128 + (lhalf << 3));
    #pragma unroll
    for (int ns = 0; ns < 4; ++ns) {
      #pragma unroll
      for (int j = 0; j < 8; ++j) {
        out[(rbase + j) * (size_t)NTOT + ncol + ns * 16] = acc[part][ns][j];
      }
    }
  }
}

// ---------------------------------------------------------------------------
// Pass 2 (per T-chunk): in-place scan over CHUNK_T steps, one thread per (b,h).
// Deep double-buffered pipeline: PF-step groups, loads for group g+1 issued
// before computing group g -> ~2-3MB in flight across 16384 threads, reading
// L2-resident data written by the immediately preceding GEMM chunk.
// Carry state lives in the f1/f2 slots of d_out between chunk launches.
// ---------------------------------------------------------------------------
__global__ __launch_bounds__(256)
void scan_chunk_kernel(const float* __restrict__ h1,
                       const float* __restrict__ h2,
                       const float* __restrict__ w1,
                       const float* __restrict__ w2,
                       float* out, int t0) {
  const int tid = blockIdx.x * 256 + threadIdx.x;   // 0..B*H-1
  const int h   = tid & (HH - 1);

  float* fout = out + (size_t)TT * BB * NTOT;       // f1 | f2 carry slots
  float p1, p2;
  if (t0 == 0) { p1 = h1[tid]; p2 = h2[tid]; }
  else         { p1 = fout[tid]; p2 = fout[BB * HH + tid]; }

  const float c1 = w1[h];
  const float c2 = w2[h];

  const size_t st = (size_t)BB * NTOT;              // elems per timestep
  float* ys = out + (size_t)t0 * st + (size_t)(tid >> 8) * NTOT + h;

  float a1[PF], a2[PF], b1[PF], b2[PF];

  #pragma unroll
  for (int i = 0; i < PF; ++i) {
    a1[i] = __builtin_nontemporal_load(ys + (size_t)i * st);
    a2[i] = __builtin_nontemporal_load(ys + (size_t)i * st + HH);
  }

  const int G = CHUNK_T / PF;                       // 32 groups (even)
  for (int g = 0; g < G; g += 2) {
    float* bg = ys + (size_t)g * PF * st;

    #pragma unroll
    for (int i = 0; i < PF; ++i) {                  // prefetch group g+1
      b1[i] = __builtin_nontemporal_load(bg + (size_t)(PF + i) * st);
      b2[i] = __builtin_nontemporal_load(bg + (size_t)(PF + i) * st + HH);
    }
    #pragma unroll
    for (int i = 0; i < PF; ++i) {                  // compute group g
      const float g1 = fmaf(c1, p1, fmaf(-c2, p2, a1[i]));
      const float g2 = fmaf(c2, p1, fmaf(c1, p2, a2[i]));
      p1 = fmaxf(g1, 0.0f);
      p2 = fmaxf(g2, 0.0f);
      __builtin_nontemporal_store(p1, bg + (size_t)i * st);
      __builtin_nontemporal_store(p2, bg + (size_t)i * st + HH);
    }
    if (g + 2 < G) {
      #pragma unroll
      for (int i = 0; i < PF; ++i) {                // prefetch group g+2
        a1[i] = __builtin_nontemporal_load(bg + (size_t)(2 * PF + i) * st);
        a2[i] = __builtin_nontemporal_load(bg + (size_t)(2 * PF + i) * st + HH);
      }
    }
    #pragma unroll
    for (int i = 0; i < PF; ++i) {                  // compute group g+1
      const float g1 = fmaf(c1, p1, fmaf(-c2, p2, b1[i]));
      const float g2 = fmaf(c2, p1, fmaf(c1, p2, b2[i]));
      p1 = fmaxf(g1, 0.0f);
      p2 = fmaxf(g2, 0.0f);
      __builtin_nontemporal_store(p1, bg + (size_t)(PF + i) * st);
      __builtin_nontemporal_store(p2, bg + (size_t)(PF + i) * st + HH);
    }
  }

  // Carry / final states: d_out = [ ys | f1 (B*H) | f2 (B*H) ]
  fout[tid]           = p1;
  fout[BB * HH + tid] = p2;
}

// ---------------------------------------------------------------------------
// Launch: interleave GEMM chunk / scan chunk so the scan reads L2-hot data.
// ---------------------------------------------------------------------------
extern "C" void kernel_launch(void* const* d_in, const int* in_sizes, int n_in,
                              void* d_out, int out_size, void* d_ws, size_t ws_size,
                              hipStream_t stream) {
  (void)in_sizes; (void)n_in; (void)out_size; (void)d_ws; (void)ws_size;

  const float* xs  = (const float*)d_in[0];
  const float* h1  = (const float*)d_in[1];
  const float* h2  = (const float*)d_in[2];
  const float* w1  = (const float*)d_in[3];
  const float* w2  = (const float*)d_in[4];
  const float* Wx1 = (const float*)d_in[5];
  const float* Wx2 = (const float*)d_in[6];
  float* out = (float*)d_out;

  const dim3 gGemm(NTOT / 64, (CHUNK_T * BB) / MBLK);   // (8, 128)
  for (int c = 0; c < NCHUNK; ++c) {
    proj_gemm_kernel<<<gGemm, 256, 0, stream>>>(xs, Wx1, Wx2, out,
                                                c * CHUNK_T * BB);
    scan_chunk_kernel<<<(BB * HH) / 256, 256, 0, stream>>>(h1, h2, w1, w2, out,
                                                           c * CHUNK_T);
  }
}
